// LCAOConv_9234179687172
// MI455X (gfx1250) — compile-verified, matrix operand order checked by reading, not run on hardware
//
#include <hip/hip_runtime.h>
#include <hip/hip_bf16.h>

// ---------------------------------------------------------------------------
// LCAOConv fused pipeline for MI455X (gfx1250, wave32, WMMA f16 16x16x32).
// Memory-bound (~1.1 GB traffic, ~14 GFLOP). GEMMs via v_wmma with fragments
// fed from LDS pre-swizzled into the ISA fragment layout (contiguous 16 halves
// per lane -> 2x ds_load_b128, no scalar packing). Graph phases: wave-per-item
// kernels with shfl butterflies + global f32 atomics.
// ---------------------------------------------------------------------------

typedef __attribute__((ext_vector_type(16))) _Float16 v16h;
typedef __attribute__((ext_vector_type(8)))  _Float16 v8h;
typedef __attribute__((ext_vector_type(8)))  float    v8f;

#define WMMA_F16(a, b, c) \
  __builtin_amdgcn_wmma_f32_16x16x32_f16(false, (a), false, (b), (short)0, (c), false, false)

__device__ __forceinline__ float siluf(float x)  { return x / (1.f + __expf(-x)); }
__device__ __forceinline__ float sigmf(float x)  { return 1.f / (1.f + __expf(-x)); }
__device__ __forceinline__ void lds_wave_fence() { asm volatile("s_wait_dscnt 0" ::: "memory"); }

// K-within-32 -> storage position. Maps [0..7]->[0..7], [8..15]->[16..23],
// [16..23]->[8..15], [24..31]->[24..31]. With this permutation, the 16 halves a
// lane needs for a 16-bit A/B fragment (ISA 7.12.2) are CONTIGUOUS:
//   lanes 0-15 read positions kbase+0..15, lanes 16-31 read kbase+16..31,
// and element order matches the fragment's VGPR order exactly.
__device__ __forceinline__ int perm32(int q) {
  return (q < 8) ? q : (q < 16 ? q + 8 : (q < 24 ? q - 8 : q));
}

// Load 16 contiguous halves (32B, 16B-aligned) as one fragment.
__device__ __forceinline__ v16h frag_ld(const _Float16* p) {
  v8h lo = *(const v8h*)p;
  v8h hi = *(const v8h*)(p + 8);
  return __builtin_shufflevector(lo, hi, 0, 1, 2, 3, 4, 5, 6, 7,
                                 8, 9, 10, 11, 12, 13, 14, 15);
}

// A fragment from a permuted-K staged tile (LDK halves per row).
__device__ __forceinline__ v16h a_frag(const _Float16* A, int ldk, int kbase, int lane) {
  return frag_ld(A + (lane & 15) * ldk + kbase + ((lane & 16) ? 16 : 0));
}

// B fragment from fragment-layout weight LDS: index ((ktI*NT+nt)*32+lane)*16.
__device__ __forceinline__ v16h b_frag(const _Float16* Wf, int NT, int ktI, int nt, int lane) {
  return frag_ld(Wf + (size_t)(((ktI * NT) + nt) * 32 + lane) * 16);
}

// One-time per block: swizzle a row-major [K x N] f32 weight into fragment
// layout f16.  KT = K/32 k-tiles, NT = N/16 n-tiles, ldn = N.
__device__ __forceinline__ void fill_wfrag(_Float16* dst, const float* W, int ldn,
                                           int KT, int NT, int tid, int nthreads) {
  int total = KT * NT * 512;
  for (int i = tid; i < total; i += nthreads) {
    int e    = i & 15;
    int lane = (i >> 4) & 31;
    int nt   = (i >> 9) % NT;
    int ktI  = (i >> 9) / NT;
    int K  = ktI * 32 + ((lane & 16) ? 8 : 0) + (e < 8 ? e : e + 8);
    int Nc = nt * 16 + (lane & 15);
    dst[i] = (_Float16)W[(size_t)K * ldn + Nc];
  }
}

// Stage a 16 x LDK activation tile into permuted-K layout; lane writes v8h
// chunks (perm32 maps aligned 8-blocks to aligned 8-blocks).
template <int LDK, typename F>
__device__ __forceinline__ void stage_a(_Float16* A, int lane, F&& ld) {
  for (int c = lane; c < 16 * LDK / 8; c += 32) {
    int r  = c / (LDK / 8);
    int k8 = (c % (LDK / 8)) * 8;
    v8h v;
#pragma unroll
    for (int q = 0; q < 8; ++q) v[q] = (_Float16)ld(r, k8 + q);
    *(v8h*)(A + r * LDK + (k8 & ~31) + perm32(k8 & 31)) = v;
  }
}
// 32-bit C/D 16x16: VGPR r -> row M = r + (lane>=16 ? 8:0), col N = lane&15.

// ---------------------------------------------------------------------------
// K1: xh = x @ W_nb + b_nb           [N,128] x [128,64] -> [N,64]
// ---------------------------------------------------------------------------
__global__ __launch_bounds__(128) void k_node_before(
    const float* __restrict__ x, const float* __restrict__ Wnb,
    const float* __restrict__ bnb, float* __restrict__ xh, int Nrows) {
  __shared__ alignas(32) _Float16 sW[128 * 64];
  __shared__ alignas(32) _Float16 sA[4][16 * 128];
  int tid = threadIdx.x;
  fill_wfrag(sW, Wnb, 64, 4, 4, tid, 128);
  __syncthreads();
  int wave = tid >> 5, lane = tid & 31;
  int row0 = (blockIdx.x * 4 + wave) * 16;
  if (row0 >= Nrows) return;
  _Float16* A = sA[wave];
  stage_a<128>(A, lane, [&](int r, int k) { return x[(size_t)(row0 + r) * 128 + k]; });
  lds_wave_fence();
  v8f acc[4] = {};
#pragma unroll
  for (int ktI = 0; ktI < 4; ++ktI) {
    v16h af = a_frag(A, 128, ktI * 32, lane);
#pragma unroll
    for (int nt = 0; nt < 4; ++nt)
      acc[nt] = WMMA_F16(af, b_frag(sW, 4, ktI, nt, lane), acc[nt]);
  }
  int mh = (lane & 16) ? 8 : 0, n = lane & 15;
#pragma unroll
  for (int nt = 0; nt < 4; ++nt)
#pragma unroll
    for (int r = 0; r < 8; ++r)
      xh[(size_t)(row0 + r + mh) * 64 + nt * 16 + n] = acc[nt][r] + bnb[nt * 16 + n];
}

// ---------------------------------------------------------------------------
// K2 (dominant): c = silu(silu(cji)@W_c1)@W_c2 over R = E*D = 480k rows.
//   cols [0,64)   -> cji_h (raw f32)
//   cols [64,128) -> ckj, L2-normalized per row (register-level butterfly)
// ---------------------------------------------------------------------------
__global__ __launch_bounds__(128) void k_coeffs(
    const float* __restrict__ cji, const float* __restrict__ Wc1,
    const float* __restrict__ Wc2, float* __restrict__ cjih,
    float* __restrict__ ckjn, int Rrows) {
  __shared__ alignas(32) _Float16 sW1[64 * 64];
  __shared__ alignas(32) _Float16 sW2[64 * 128];
  __shared__ alignas(32) _Float16 sA[4][16 * 64];
  __shared__ alignas(32) _Float16 sM[4][16 * 64];
  int tid = threadIdx.x;
  fill_wfrag(sW1, Wc1, 64, 2, 4, tid, 128);
  fill_wfrag(sW2, Wc2, 128, 2, 8, tid, 128);
  __syncthreads();
  int wave = tid >> 5, lane = tid & 31;
  int row0 = (blockIdx.x * 4 + wave) * 16;
  if (row0 >= Rrows) return;
  _Float16* A = sA[wave];
  _Float16* M = sM[wave];
  stage_a<64>(A, lane, [&](int r, int k) { return siluf(cji[(size_t)(row0 + r) * 64 + k]); });
  lds_wave_fence();
  // GEMM1: [16,64] x [64,64]
  v8f acc1[4] = {};
#pragma unroll
  for (int ktI = 0; ktI < 2; ++ktI) {
    v16h af = a_frag(A, 64, ktI * 32, lane);
#pragma unroll
    for (int nt = 0; nt < 4; ++nt)
      acc1[nt] = WMMA_F16(af, b_frag(sW1, 4, ktI, nt, lane), acc1[nt]);
  }
  int mh = (lane & 16) ? 8 : 0, n = lane & 15;
  // mid = silu(acc1) -> restage (permuted columns = next GEMM's K)
#pragma unroll
  for (int nt = 0; nt < 4; ++nt) {
    int c  = nt * 16 + n;
    int cs = (c & ~31) + perm32(c & 31);
#pragma unroll
    for (int r = 0; r < 8; ++r)
      M[(r + mh) * 64 + cs] = (_Float16)siluf(acc1[nt][r]);
  }
  lds_wave_fence();
  // GEMM2: [16,64] x [64,128]
  v8f acc2[8] = {};
#pragma unroll
  for (int ktI = 0; ktI < 2; ++ktI) {
    v16h af = a_frag(M, 64, ktI * 32, lane);
#pragma unroll
    for (int nt = 0; nt < 8; ++nt)
      acc2[nt] = WMMA_F16(af, b_frag(sW2, 8, ktI, nt, lane), acc2[nt]);
  }
  // cols 0..63 -> cji_h
#pragma unroll
  for (int nt = 0; nt < 4; ++nt)
#pragma unroll
    for (int r = 0; r < 8; ++r)
      cjih[(size_t)(row0 + r + mh) * 64 + nt * 16 + n] = acc2[nt][r];
  // cols 64..127 -> ckj, normalize per row directly in D-register layout:
  // lanes 0-15 hold row r+0..?: VGPR r = row (r+mh); 16-lane butterfly (masks
  // 8..1) reduces across the 16 columns held by each lane half; summing the 4
  // column tiles gives the full 64-col sum-of-squares for that row.
#pragma unroll
  for (int r = 0; r < 8; ++r) {
    float ss = 0.f;
#pragma unroll
    for (int nt = 4; nt < 8; ++nt) { float v = acc2[nt][r]; ss += v * v; }
#pragma unroll
    for (int m = 8; m >= 1; m >>= 1) ss += __shfl_xor(ss, m, 32);
    float inv = 1.f / fmaxf(sqrtf(ss), 1e-12f);
#pragma unroll
    for (int nt = 4; nt < 8; ++nt)
      ckjn[(size_t)(row0 + r + mh) * 64 + (nt - 4) * 16 + n] = acc2[nt][r] * inv;
  }
}

// ---------------------------------------------------------------------------
// K3: wave-per-triplet. tbo = normalize(sum_d (shbs*robs)[d]*ckj_norm[kj,d,:])
//     * sigmoid(xh[k]); atomic scatter into three_body_w[ji].
// ---------------------------------------------------------------------------
__global__ __launch_bounds__(256) void k_triplet(
    const float* __restrict__ ckjn, const float* __restrict__ xh,
    const float* __restrict__ robs, const float* __restrict__ shbs,
    const int* __restrict__ tri_k, const int* __restrict__ e_kj,
    const int* __restrict__ e_ji, float* __restrict__ tbw, int T) {
  int t = (blockIdx.x * blockDim.x + threadIdx.x) >> 5;
  int lane = threadIdx.x & 31;
  if (t >= T) return;
  int kj = e_kj[t], k = tri_k[t], ji = e_ji[t];
  const float* cb = ckjn + (size_t)kj * (12 * 64);
  float ax = 0.f, ay = 0.f;   // each lane owns channels 2*lane, 2*lane+1
#pragma unroll
  for (int d = 0; d < 12; ++d) {
    float w = shbs[(size_t)t * 12 + d] * robs[(size_t)kj * 12 + d];
    float2 c2 = *(const float2*)(cb + d * 64 + lane * 2);
    ax += w * c2.x;
    ay += w * c2.y;
  }
  float ss = ax * ax + ay * ay;
#pragma unroll
  for (int m = 16; m >= 1; m >>= 1) ss += __shfl_xor(ss, m, 32);
  float inv = 1.f / fmaxf(sqrtf(ss), 1e-12f);
  float2 xv = *(const float2*)(xh + (size_t)k * 64 + lane * 2);
  float* dst = tbw + (size_t)ji * 64 + lane * 2;
  atomicAdd(dst,     ax * inv * sigmf(xv.x));
  atomicAdd(dst + 1, ay * inv * sigmf(xv.y));
}

// ---------------------------------------------------------------------------
// K4a: edge MLP. h2 = silu(silu([xh_i|xh_j])@W_n1+b_n1)@W_n2+b_n2  -> [E,64]
// ---------------------------------------------------------------------------
__global__ __launch_bounds__(128) void k_edge_mlp(
    const float* __restrict__ xh, const int* __restrict__ idx_i,
    const int* __restrict__ idx_j, const float* __restrict__ Wn1,
    const float* __restrict__ bn1, const float* __restrict__ Wn2,
    const float* __restrict__ bn2, float* __restrict__ h2, int E) {
  __shared__ alignas(32) _Float16 sW1[128 * 64];
  __shared__ alignas(32) _Float16 sW2[64 * 64];
  __shared__ alignas(32) _Float16 sA[4][16 * 128];
  __shared__ alignas(32) _Float16 sM[4][16 * 64];
  int tid = threadIdx.x;
  fill_wfrag(sW1, Wn1, 64, 4, 4, tid, 128);
  fill_wfrag(sW2, Wn2, 64, 2, 4, tid, 128);
  __syncthreads();
  int wave = tid >> 5, lane = tid & 31;
  int row0 = (blockIdx.x * 4 + wave) * 16;
  if (row0 >= E) return;
  _Float16* A = sA[wave];
  _Float16* M = sM[wave];
  stage_a<128>(A, lane, [&](int r, int k) {
    int node = (k < 64) ? idx_i[row0 + r] : idx_j[row0 + r];
    return siluf(xh[(size_t)node * 64 + (k & 63)]);
  });
  lds_wave_fence();
  v8f acc1[4] = {};
#pragma unroll
  for (int ktI = 0; ktI < 4; ++ktI) {
    v16h af = a_frag(A, 128, ktI * 32, lane);
#pragma unroll
    for (int nt = 0; nt < 4; ++nt)
      acc1[nt] = WMMA_F16(af, b_frag(sW1, 4, ktI, nt, lane), acc1[nt]);
  }
  int mh = (lane & 16) ? 8 : 0, n = lane & 15;
#pragma unroll
  for (int nt = 0; nt < 4; ++nt) {
    int c  = nt * 16 + n;
    int cs = (c & ~31) + perm32(c & 31);
    float b = bn1[c];
#pragma unroll
    for (int r = 0; r < 8; ++r)
      M[(r + mh) * 64 + cs] = (_Float16)siluf(acc1[nt][r] + b);
  }
  lds_wave_fence();
  v8f acc2[4] = {};
#pragma unroll
  for (int ktI = 0; ktI < 2; ++ktI) {
    v16h af = a_frag(M, 64, ktI * 32, lane);
#pragma unroll
    for (int nt = 0; nt < 4; ++nt)
      acc2[nt] = WMMA_F16(af, b_frag(sW2, 4, ktI, nt, lane), acc2[nt]);
  }
#pragma unroll
  for (int nt = 0; nt < 4; ++nt)
#pragma unroll
    for (int r = 0; r < 8; ++r)
      h2[(size_t)(row0 + r + mh) * 64 + nt * 16 + n] = acc2[nt][r] + bn2[nt * 16 + n];
}

// ---------------------------------------------------------------------------
// K4b: wave-per-edge. Per-d normalize of cji_h*(1+tbw), radial contraction,
//      final normalize, msg = lcao_w * h2, atomic scatter to agg[idx_i].
// ---------------------------------------------------------------------------
__global__ __launch_bounds__(256) void k_edge_combine(
    const float* __restrict__ cjih, const float* __restrict__ tbw,
    const float* __restrict__ robs, const float* __restrict__ h2,
    const int* __restrict__ idx_i, float* __restrict__ agg, int E) {
  int e = (blockIdx.x * blockDim.x + threadIdx.x) >> 5;
  int lane = threadIdx.x & 31;
  if (e >= E) return;
  float2 t2 = *(const float2*)(tbw + (size_t)e * 64 + lane * 2);
  const float* cb = cjih + (size_t)e * (12 * 64);
  float lx = 0.f, ly = 0.f;
#pragma unroll
  for (int d = 0; d < 12; ++d) {
    float2 c2 = *(const float2*)(cb + d * 64 + lane * 2);
    float vx = c2.x * (1.f + t2.x);
    float vy = c2.y * (1.f + t2.y);
    float ss = vx * vx + vy * vy;
#pragma unroll
    for (int m = 16; m >= 1; m >>= 1) ss += __shfl_xor(ss, m, 32);
    float w = robs[(size_t)e * 12 + d] / fmaxf(sqrtf(ss), 1e-12f);
    lx += w * vx;
    ly += w * vy;
  }
  float ss = lx * lx + ly * ly;
#pragma unroll
  for (int m = 16; m >= 1; m >>= 1) ss += __shfl_xor(ss, m, 32);
  float inv = 1.f / fmaxf(sqrtf(ss), 1e-12f);
  float2 hh = *(const float2*)(h2 + (size_t)e * 64 + lane * 2);
  float* dst = agg + (size_t)idx_i[e] * 64 + lane * 2;
  atomicAdd(dst,     lx * inv * hh.x);
  atomicAdd(dst + 1, ly * inv * hh.y);
}

// ---------------------------------------------------------------------------
// K5: out = x + agg @ W_na           [N,64] x [64,128] -> [N,128]
// ---------------------------------------------------------------------------
__global__ __launch_bounds__(128) void k_node_after(
    const float* __restrict__ agg, const float* __restrict__ Wna,
    const float* __restrict__ x, float* __restrict__ out, int Nrows) {
  __shared__ alignas(32) _Float16 sW[64 * 128];
  __shared__ alignas(32) _Float16 sA[4][16 * 64];
  int tid = threadIdx.x;
  fill_wfrag(sW, Wna, 128, 2, 8, tid, 128);
  __syncthreads();
  int wave = tid >> 5, lane = tid & 31;
  int row0 = (blockIdx.x * 4 + wave) * 16;
  if (row0 >= Nrows) return;
  _Float16* A = sA[wave];
  stage_a<64>(A, lane, [&](int r, int k) { return agg[(size_t)(row0 + r) * 64 + k]; });
  lds_wave_fence();
  v8f acc[8] = {};
#pragma unroll
  for (int ktI = 0; ktI < 2; ++ktI) {
    v16h af = a_frag(A, 64, ktI * 32, lane);
#pragma unroll
    for (int nt = 0; nt < 8; ++nt)
      acc[nt] = WMMA_F16(af, b_frag(sW, 8, ktI, nt, lane), acc[nt]);
  }
  int mh = (lane & 16) ? 8 : 0, n = lane & 15;
#pragma unroll
  for (int nt = 0; nt < 8; ++nt)
#pragma unroll
    for (int r = 0; r < 8; ++r) {
      size_t o = (size_t)(row0 + r + mh) * 128 + nt * 16 + n;
      out[o] = x[o] + acc[nt][r];
    }
}

// ---------------------------------------------------------------------------
extern "C" void kernel_launch(void* const* d_in, const int* in_sizes, int n_in,
                              void* d_out, int out_size, void* d_ws, size_t ws_size,
                              hipStream_t stream) {
  const float* x    = (const float*)d_in[0];
  const float* cji  = (const float*)d_in[1];
  const float* robs = (const float*)d_in[2];
  const float* shbs = (const float*)d_in[3];
  const int*   idxi = (const int*)d_in[4];
  const int*   idxj = (const int*)d_in[5];
  const int*   trik = (const int*)d_in[6];
  const int*   ekj  = (const int*)d_in[7];
  const int*   eji  = (const int*)d_in[8];
  const float* Wc1  = (const float*)d_in[9];
  const float* Wc2  = (const float*)d_in[10];
  const float* Wnb  = (const float*)d_in[11];
  const float* bnb  = (const float*)d_in[12];
  const float* Wn1  = (const float*)d_in[13];
  const float* bn1  = (const float*)d_in[14];
  const float* Wn2  = (const float*)d_in[15];
  const float* bn2  = (const float*)d_in[16];
  const float* Wna  = (const float*)d_in[17];
  float* out = (float*)d_out;
  (void)n_in; (void)out_size; (void)ws_size;

  const int N = in_sizes[0] / 128;   // x: [N,128]
  const int E = in_sizes[2] / 12;    // robs: [E,12]
  const int T = in_sizes[3] / 12;    // shbs: [T,12]
  const int R = E * 12;              // flattened (e,d) rows

  // Workspace partition (f32): xh | cji_h | ckj_norm | tbw | h2 | agg
  float* ws   = (float*)d_ws;
  float* xh   = ws;                 size_t o = (size_t)N * 64;
  float* cjih = ws + o;             o += (size_t)R * 64;
  float* ckjn = ws + o;             o += (size_t)R * 64;
  float* tbw  = ws + o;             o += (size_t)E * 64;
  float* h2   = ws + o;             o += (size_t)E * 64;
  float* agg  = ws + o;             o += (size_t)N * 64;

  hipMemsetAsync(tbw, 0, (size_t)E * 64 * sizeof(float), stream);
  hipMemsetAsync(agg, 0, (size_t)N * 64 * sizeof(float), stream);

  const int tilesN = (N + 15) / 16;
  const int tilesR = (R + 15) / 16;
  const int tilesE = (E + 15) / 16;

  k_node_before<<<(tilesN + 3) / 4, 128, 0, stream>>>(x, Wnb, bnb, xh, N);
  k_coeffs<<<(tilesR + 3) / 4, 128, 0, stream>>>(cji, Wc1, Wc2, cjih, ckjn, R);
  k_triplet<<<(T + 7) / 8, 256, 0, stream>>>(ckjn, xh, robs, shbs, trik, ekj, eji, tbw, T);
  k_edge_mlp<<<(tilesE + 3) / 4, 128, 0, stream>>>(xh, idxi, idxj, Wn1, bn1, Wn2, bn2, h2, E);
  k_edge_combine<<<(E + 7) / 8, 256, 0, stream>>>(cjih, tbw, robs, h2, idxi, agg, E);
  k_node_after<<<(tilesN + 3) / 4, 128, 0, stream>>>(agg, Wna, x, out, N);
}
